// ChaoticLSTM_5549097747161
// MI455X (gfx1250) — matrix operand order repaired and checked by reading, not compiled, not run on hardware
//
#include <hip/hip_runtime.h>
#include <hip/hip_bf16.h>

typedef __attribute__((ext_vector_type(16))) _Float16 v16h;
typedef __attribute__((ext_vector_type(8)))  _Float16 v8h;
typedef __attribute__((ext_vector_type(8)))  float    v8f;

#define SEQLEN 2048
#define NBATCH 32
#define HID    128
#define GDIM   512   // 4*HID

__device__ __forceinline__ float fast_sigmoid(float x) {
    // 1 / (1 + exp(-x)) = rcp(1 + exp2(-x*log2(e)))
    return __builtin_amdgcn_rcpf(1.0f + __builtin_amdgcn_exp2f(-1.44269504089f * x));
}
__device__ __forceinline__ float fast_tanh(float x) {
    // tanh(x) = 1 - 2/(1 + exp(2x)) = 1 - 2*rcp(1 + exp2(2x*log2(e)))
    return 1.0f - 2.0f * __builtin_amdgcn_rcpf(1.0f + __builtin_amdgcn_exp2f(2.88539008178f * x));
}

// ---------------------------------------------------------------------------
// Phase 1: XW[t][g][b] = (x[b,t,:] @ Wi)[g] + B[g], stored as f16.
// M = t*32 + b flattened; one block per 16-row M-tile; 8 waves x 4 n-tiles.
// Output layout [t][512][32] so phase 2 reads accumulator-init as v8h (16B).
// ---------------------------------------------------------------------------
__global__ __launch_bounds__(256) void lstm_xw_gemm(
        const float* __restrict__ x, const float* __restrict__ Wi,
        const float* __restrict__ Bias, _Float16* __restrict__ XW) {
    const int mt   = blockIdx.x;        // 0..4095
    const int t    = mt >> 1;
    const int b0   = (mt & 1) * 16;
    const int lane = threadIdx.x & 31;
    const int wid  = threadIdx.x >> 5;  // 0..7
    const int l15  = lane & 15;
    const bool hic = lane >= 16;

    // A tiles: rows = batch, K = input(128). 16-bit A layout:
    // element e holds K = e + (e<8?0:8) + (hi-half?8:0), M = lane&15.
    v16h a[4];
#pragma unroll
    for (int j = 0; j < 4; ++j) {
        const int brow = b0 + l15;
        const float* xp = x + ((size_t)brow * SEQLEN + t) * 128 + 32 * j + (hic ? 8 : 0);
        v16h av;
#pragma unroll
        for (int e = 0; e < 16; ++e) {
            const int k = e + (e < 8 ? 0 : 8);
            av[e] = (_Float16)xp[k];
        }
        a[j] = av;
    }

#pragma unroll
    for (int gi = 0; gi < 4; ++gi) {
        const int gt  = wid + 8 * gi;        // n-tile 0..31
        const int col = 16 * gt + l15;
        const float bv = Bias[col];
        v8f acc;
#pragma unroll
        for (int r = 0; r < 8; ++r) acc[r] = bv;   // fold bias into XW
#pragma unroll
        for (int j = 0; j < 4; ++j) {
            // B tile 32x16: element e holds K = e + (e<8?0:8) + (hi?8:0), N = lane&15
            const float* wp = Wi + (size_t)(32 * j + (hic ? 8 : 0)) * GDIM + col;
            v16h wv;
#pragma unroll
            for (int e = 0; e < 16; ++e) wv[e] = (_Float16)wp[(size_t)(e + (e < 8 ? 0 : 8)) * GDIM];
            acc = __builtin_amdgcn_wmma_f32_16x16x32_f16(
                      false, a[j], false, wv, (short)0, acc, false, false);
        }
        v8h o;
#pragma unroll
        for (int r = 0; r < 8; ++r) o[r] = (_Float16)acc[r];
        const int bbase = b0 + (hic ? 8 : 0);
        *(v8h*)(XW + ((size_t)t * GDIM + col) * NBATCH + bbase) = o;
    }
}

// ---------------------------------------------------------------------------
// Phase 2: sequential recurrence. One block per direction, 16 waves.
// Wave (m,n): gate tiles {n, n+8, n+16, n+24} for rows 16m..16m+15.
// Wh lives in VGPRs (16 x v16h); c lives in VGPRs; h cycles through LDS f16.
// ---------------------------------------------------------------------------
__global__ __launch_bounds__(512) void lstm_recurrence(
        const _Float16* __restrict__ XW, const float* __restrict__ Wh,
        float* __restrict__ out, float* __restrict__ carry) {
    const int dir  = blockIdx.x;          // 0 = forward, 1 = backward
    const int lane = threadIdx.x & 31;
    const int wid  = threadIdx.x >> 5;    // 0..15
    const int m    = wid & 1;
    const int n    = wid >> 1;            // hidden col tile 0..7
    const int l15  = lane & 15;
    const bool hic = lane >= 16;

    __shared__ _Float16 hbuf[NBATCH * HID];   // 8 KB

    for (int i = threadIdx.x; i < NBATCH * HID; i += 512) hbuf[i] = (_Float16)0.0f;

    // Preload Wh B-tiles into registers: w[gate][kstep]
    v16h w[4][4];
#pragma unroll
    for (int gi = 0; gi < 4; ++gi) {
        const int col = 16 * (n + 8 * gi) + l15;
#pragma unroll
        for (int j = 0; j < 4; ++j) {
            const float* wp = Wh + (size_t)(32 * j + (hic ? 8 : 0)) * GDIM + col;
            v16h wv;
#pragma unroll
            for (int e = 0; e < 16; ++e) wv[e] = (_Float16)wp[(size_t)(e + (e < 8 ? 0 : 8)) * GDIM];
            w[gi][j] = wv;
        }
    }
    __syncthreads();

    v8f c, hreg;
#pragma unroll
    for (int r = 0; r < 8; ++r) { c[r] = 0.0f; hreg[r] = 0.0f; }

    const int bbase = 16 * m + (hic ? 8 : 0);   // C-layout batch base
    const int hrow  = 16 * m + l15;             // A-layout batch row

    // Preload XW for step 0 (prefetched one step ahead thereafter)
    v8h xw[4];
    {
        const int xt = dir ? (SEQLEN - 1) : 0;
#pragma unroll
        for (int gi = 0; gi < 4; ++gi) {
            const int col = 16 * (n + 8 * gi) + l15;
            xw[gi] = *(const v8h*)(XW + ((size_t)xt * GDIM + col) * NBATCH + bbase);
        }
    }

    for (int s = 0; s < SEQLEN; ++s) {
        v8f acc[4];
#pragma unroll
        for (int gi = 0; gi < 4; ++gi)
#pragma unroll
            for (int r = 0; r < 8; ++r) acc[gi][r] = (float)xw[gi][r];

        // Prefetch next step's x@Wi contribution (independent of the scan dep)
        if (s + 1 < SEQLEN) {
            const int xt = dir ? (SEQLEN - 2 - s) : (s + 1);
#pragma unroll
            for (int gi = 0; gi < 4; ++gi) {
                const int col = 16 * (n + 8 * gi) + l15;
                xw[gi] = *(const v8h*)(XW + ((size_t)xt * GDIM + col) * NBATCH + bbase);
            }
        }

        // gates += h @ Wh : 16 WMMAs per wave
#pragma unroll
        for (int j = 0; j < 4; ++j) {
            const _Float16* hp = hbuf + hrow * HID + 32 * j + (hic ? 8 : 0);
            const v8h lo = *(const v8h*)(hp);
            const v8h hh = *(const v8h*)(hp + 16);
            v16h av;
#pragma unroll
            for (int e = 0; e < 8; ++e) { av[e] = lo[e]; av[8 + e] = hh[e]; }
#pragma unroll
            for (int gi = 0; gi < 4; ++gi)
                acc[gi] = __builtin_amdgcn_wmma_f32_16x16x32_f16(
                              false, av, false, w[gi][j], (short)0, acc[gi], false, false);
        }
        __syncthreads();   // all hbuf reads done before overwrite

        const int out_col = dir * HID + 16 * n + l15;
#pragma unroll
        for (int r = 0; r < 8; ++r) {
            const float ig = fast_sigmoid(acc[0][r]);
            const float fg = fast_sigmoid(acc[1][r]);
            const float gg = fast_tanh(acc[2][r]);
            const float og = fast_sigmoid(acc[3][r]);
            const float cv = fg * c[r] + ig * gg;
            c[r] = cv;
            const float hv = og * fast_tanh(cv);
            hreg[r] = hv;
            const int b = bbase + r;
            hbuf[b * HID + 16 * n + l15] = (_Float16)hv;
            out[((size_t)b * SEQLEN + s) * (2 * HID) + out_col] = hv;
        }
        __syncthreads();   // h(s) visible to all waves for step s+1
    }

    // carry: (ht, ct, hinvt, cinvt), each [32][128]
    float* cb = carry + (size_t)dir * 2 * NBATCH * HID;
#pragma unroll
    for (int r = 0; r < 8; ++r) {
        const int b = bbase + r;
        cb[b * HID + 16 * n + l15] = hreg[r];
        cb[NBATCH * HID + b * HID + 16 * n + l15] = c[r];
    }
}

extern "C" void kernel_launch(void* const* d_in, const int* in_sizes, int n_in,
                              void* d_out, int out_size, void* d_ws, size_t ws_size,
                              hipStream_t stream) {
    const float* x  = (const float*)d_in[0];
    const float* Wi = (const float*)d_in[1];
    const float* Wh = (const float*)d_in[2];
    const float* B  = (const float*)d_in[3];
    float* out   = (float*)d_out;
    float* carry = out + (size_t)NBATCH * SEQLEN * 2 * HID;

    // Workspace: XW = 2048*512*32 f16 = 64 MiB (ws assumed >= that)
    _Float16* XW = (_Float16*)d_ws;

    lstm_xw_gemm<<<4096, 256, 0, stream>>>(x, Wi, B, XW);
    lstm_recurrence<<<2, 512, 0, stream>>>(XW, Wh, out, carry);
}
